// AttentionWithTime_79018808312150
// MI455X (gfx1250) — matrix operand order repaired and checked by reading, not compiled, not run on hardware
//
#include <hip/hip_runtime.h>
#include <hip/hip_bf16.h>
#include <math.h>

// ---------------------------------------------------------------------------
// Types for gfx1250 WMMA
// ---------------------------------------------------------------------------
typedef __bf16 bf16_t;
typedef bf16_t v16bf __attribute__((ext_vector_type(16)));
typedef bf16_t v8bf  __attribute__((ext_vector_type(8)));
typedef float  v8f   __attribute__((ext_vector_type(8)));

#define IN_C     512
#define HEAD_C   64
#define EXPAND_C 2048
#define OUT_CH   512
#define TIME_C   256
#define NHEAD    8
#define QKVD     5120     // 512 q + 512 k + 4096 v
#define BATCH    4
#define SEQ      2048
#define ROWS     (BATCH*SEQ)   // 8192
#define LN_EPS   1e-5f

__device__ __forceinline__ v16bf ldfrag(const bf16_t* p0, const bf16_t* p1) {
    v8bf lo = *(const v8bf*)p0;
    v8bf hi = *(const v8bf*)p1;
    return __builtin_shufflevector(lo, hi, 0,1,2,3,4,5,6,7,8,9,10,11,12,13,14,15);
}

__device__ __forceinline__ v8f wmma_bf16(v16bf a, v16bf b, v8f c) {
    // D = A(16x32) * B(32x16) + C, f32 accumulate
    return __builtin_amdgcn_wmma_f32_16x16x32_bf16(false, a, false, b, (short)0, c,
                                                   false, false);
}

__device__ __forceinline__ float gelu_exact(float v) {
    return 0.5f * v * (1.0f + erff(v * 0.70710678118654752f));
}

// ---------------------------------------------------------------------------
// CDNA5 async DMA: global -> LDS, 16B per lane, tracked by ASYNCcnt.
// Generic address of a __shared__ object carries the LDS offset in bits[31:0]
// (aperture bits live in [63:32]); pass the truncated pointer as dest VGPR.
// ---------------------------------------------------------------------------
__device__ __forceinline__ unsigned lds_off(const void* lptr) {
    return (unsigned)(size_t)lptr;
}
__device__ __forceinline__ void async_load_b128(const void* gptr, unsigned lds) {
    asm volatile("global_load_async_to_lds_b128 %0, %1, off"
                 :: "v"(lds), "v"(gptr) : "memory");
}
__device__ __forceinline__ void wait_async() {
    asm volatile("s_wait_asynccnt 0" ::: "memory");
}

// ---------------------------------------------------------------------------
// Weight convert + transpose: W (K x N, f32 row-major) -> Wt (N x K, bf16)
// ---------------------------------------------------------------------------
__global__ __launch_bounds__(256) void convert_wt(const float* __restrict__ W,
                                                  bf16_t* __restrict__ Wt,
                                                  int K, int N) {
    size_t idx = (size_t)blockIdx.x * 256 + threadIdx.x;
    size_t total = (size_t)K * (size_t)N;
    if (idx >= total) return;
    size_t n = idx / (size_t)K;
    size_t k = idx - n * (size_t)K;
    Wt[idx] = (bf16_t)W[k * (size_t)N + n];
}

// ---------------------------------------------------------------------------
// LayerNorm over 512 (x) -> bf16
// ---------------------------------------------------------------------------
__global__ __launch_bounds__(256) void ln1_kernel(const float* __restrict__ x,
                                                  const float* __restrict__ g,
                                                  const float* __restrict__ bt,
                                                  bf16_t* __restrict__ y) {
    __shared__ float red[256];
    const int row = blockIdx.x, tid = threadIdx.x;
    const float* xr = x + (size_t)row * IN_C;
    float v0 = xr[tid], v1 = xr[tid + 256];
    red[tid] = v0 + v1;
    __syncthreads();
    for (int st = 128; st > 0; st >>= 1) {
        if (tid < st) red[tid] += red[tid + st];
        __syncthreads();
    }
    float mu = red[0] * (1.0f / IN_C);
    __syncthreads();
    float d0 = v0 - mu, d1 = v1 - mu;
    red[tid] = d0 * d0 + d1 * d1;
    __syncthreads();
    for (int st = 128; st > 0; st >>= 1) {
        if (tid < st) red[tid] += red[tid + st];
        __syncthreads();
    }
    float rstd = rsqrtf(red[0] * (1.0f / IN_C) + LN_EPS);
    bf16_t* yr = y + (size_t)row * IN_C;
    yr[tid]       = (bf16_t)(d0 * rstd * g[tid]       + bt[tid]);
    yr[tid + 256] = (bf16_t)(d1 * rstd * g[tid + 256] + bt[tid + 256]);
}

// ---------------------------------------------------------------------------
// LayerNorm over concat(x1[512], tt[256]) = 768 -> bf16
// ---------------------------------------------------------------------------
__global__ __launch_bounds__(256) void ln2_kernel(const float* __restrict__ x1,
                                                  const float* __restrict__ tt,
                                                  const float* __restrict__ g,
                                                  const float* __restrict__ bt,
                                                  bf16_t* __restrict__ y) {
    __shared__ float red[256];
    const int row = blockIdx.x, tid = threadIdx.x;
    const int b = row >> 11;   // row / SEQ
    float v0 = x1[(size_t)row * IN_C + tid];
    float v1 = x1[(size_t)row * IN_C + tid + 256];
    float v2 = tt[(size_t)b * TIME_C + tid];
    red[tid] = v0 + v1 + v2;
    __syncthreads();
    for (int st = 128; st > 0; st >>= 1) {
        if (tid < st) red[tid] += red[tid + st];
        __syncthreads();
    }
    float mu = red[0] * (1.0f / 768.0f);
    __syncthreads();
    float d0 = v0 - mu, d1 = v1 - mu, d2 = v2 - mu;
    red[tid] = d0 * d0 + d1 * d1 + d2 * d2;
    __syncthreads();
    for (int st = 128; st > 0; st >>= 1) {
        if (tid < st) red[tid] += red[tid + st];
        __syncthreads();
    }
    float rstd = rsqrtf(red[0] * (1.0f / 768.0f) + LN_EPS);
    bf16_t* yr = y + (size_t)row * 768;
    yr[tid]       = (bf16_t)(d0 * rstd * g[tid]       + bt[tid]);
    yr[tid + 256] = (bf16_t)(d1 * rstd * g[tid + 256] + bt[tid + 256]);
    yr[tid + 512] = (bf16_t)(d2 * rstd * g[tid + 512] + bt[tid + 512]);
}

// ---------------------------------------------------------------------------
// Tiny time MLP: tt[b,j] = t[b,:] @ time_w[:,j] + time_b[j]   (4 x 256)
// ---------------------------------------------------------------------------
__global__ __launch_bounds__(256) void time_kernel(const float* __restrict__ t,
                                                   const float* __restrict__ w,
                                                   const float* __restrict__ bias,
                                                   float* __restrict__ tt) {
    const int b = blockIdx.x, j = threadIdx.x;
    float acc = bias[j];
    const float* tr = t + (size_t)b * TIME_C;
    for (int i = 0; i < TIME_C; ++i)
        acc += tr[i] * w[(size_t)i * TIME_C + j];
    tt[(size_t)b * TIME_C + j] = acc;
}

// ---------------------------------------------------------------------------
// Generic bf16 WMMA GEMM:  C(MxN) = A(MxK) @ Bt(NxK)^T + bias [+GELU] [+resid]
// 128x128 WG tile, 8 waves (4x2), 32x64 per wave, K-step 32.
// Double-buffered LDS staged by global_load_async_to_lds_b128; all staging
// addresses are pre-computed and advanced by +64B/step (no per-iter muls).
// ---------------------------------------------------------------------------
template <int ACT, int RES, int OF32>
__global__ __launch_bounds__(256) void gemm_bf16(const bf16_t* __restrict__ A,
                                                 const bf16_t* __restrict__ Bt,
                                                 const float* __restrict__ bias,
                                                 const float* __restrict__ resid,
                                                 void* __restrict__ Cout,
                                                 int M, int N, int K) {
    __shared__ bf16_t As[2][128 * 40];   // 32 K-elems + 8 pad per row
    __shared__ bf16_t Bs[2][128 * 40];

    const int tid = threadIdx.x, lane = tid & 31, wid = tid >> 5;
    const int hb = lane >> 4, mm = lane & 15;
    const int wr = wid >> 1, wc = wid & 1;
    const int m0 = blockIdx.y * 128, n0 = blockIdx.x * 128;

    // two 16B chunks per thread per matrix per k-step
    const int r0 = tid >> 2, r1 = r0 + 64, cc = (tid & 3) * 8;

    // incrementing global pointers (+32 elems = 64B per k-step)
    const bf16_t* gA0 = &A[(size_t)(m0 + r0) * K + cc];
    const bf16_t* gA1 = &A[(size_t)(m0 + r1) * K + cc];
    const bf16_t* gB0 = &Bt[(size_t)(n0 + r0) * K + cc];
    const bf16_t* gB1 = &Bt[(size_t)(n0 + r1) * K + cc];
    // fixed LDS destinations per buffer
    unsigned lA0[2], lA1[2], lB0[2], lB1[2];
#pragma unroll
    for (int bf = 0; bf < 2; ++bf) {
        lA0[bf] = lds_off(&As[bf][r0 * 40 + cc]);
        lA1[bf] = lds_off(&As[bf][r1 * 40 + cc]);
        lB0[bf] = lds_off(&Bs[bf][r0 * 40 + cc]);
        lB1[bf] = lds_off(&Bs[bf][r1 * 40 + cc]);
    }

    v8f acc[2][4];
#pragma unroll
    for (int r = 0; r < 2; ++r)
#pragma unroll
        for (int c = 0; c < 4; ++c) {
            v8f z = {0.f, 0.f, 0.f, 0.f, 0.f, 0.f, 0.f, 0.f};
            acc[r][c] = z;
        }

    auto stage = [&](int buf) {
        async_load_b128(gA0, lA0[buf]); gA0 += 32;
        async_load_b128(gA1, lA1[buf]); gA1 += 32;
        async_load_b128(gB0, lB0[buf]); gB0 += 32;
        async_load_b128(gB1, lB1[buf]); gB1 += 32;
    };

    const int nk = K >> 5;
    stage(0);
    wait_async();
    __syncthreads();

    for (int kt = 0; kt < nk; ++kt) {
        const int buf = kt & 1;
        if (kt + 1 < nk) stage(buf ^ 1);   // DMA next tile while computing

        v16bf af[2], bfr[4];
#pragma unroll
        for (int r = 0; r < 2; ++r) {
            // A frag 16x32: lanes 0-15 K=0..15, lanes 16-31 K=16..31
            const bf16_t* p = &As[buf][(wr * 32 + r * 16 + mm) * 40 + hb * 16];
            af[r] = ldfrag(p, p + 8);
        }
#pragma unroll
        for (int c = 0; c < 4; ++c) {
            // B frag 32x16: lanes 0-15 K=0..15, lanes 16-31 K=16..31
            const bf16_t* p = &Bs[buf][(wc * 64 + c * 16 + mm) * 40 + hb * 16];
            bfr[c] = ldfrag(p, p + 8);
        }
#pragma unroll
        for (int r = 0; r < 2; ++r)
#pragma unroll
            for (int c = 0; c < 4; ++c)
                acc[r][c] = wmma_bf16(af[r], bfr[c], acc[r][c]);

        wait_async();        // next tile landed in LDS (no-op on last iter)
        __syncthreads();
    }

    // epilogue: C layout — VGPR i holds row i + 8*hb, col = mm
#pragma unroll
    for (int c = 0; c < 4; ++c) {
        int gn = n0 + wc * 64 + c * 16 + mm;
        float bs = bias[gn];
#pragma unroll
        for (int r = 0; r < 2; ++r) {
#pragma unroll
            for (int i = 0; i < 8; ++i) {
                int gm = m0 + wr * 32 + r * 16 + i + hb * 8;
                float v = acc[r][c][i] + bs;
                if constexpr (ACT) v = gelu_exact(v);
                if constexpr (RES) v += resid[(size_t)gm * N + gn];
                if constexpr (OF32)
                    ((float*)Cout)[(size_t)gm * N + gn] = v;
                else
                    ((bf16_t*)Cout)[(size_t)gm * N + gn] = (bf16_t)v;
            }
        }
    }
}

// ---------------------------------------------------------------------------
// Flash attention. qkv: (ROWS x 5120) bf16, cols [0,512)=q, [512,1024)=k,
// [1024,5120)=v (per-head V dim 512). One WG = (b, h, 64-query block).
// 8 waves: waves 0-3 own 16 query rows each (S + online softmax),
// all 8 waves own 64 v-columns each for P@V. Key blocks of 32.
// V tile is transposed 8x8-in-registers, then stored with ds_store_b128.
// ---------------------------------------------------------------------------
__global__ __launch_bounds__(256) void flash_attn(const bf16_t* __restrict__ qkv,
                                                  bf16_t* __restrict__ attn_out) {
    __shared__ bf16_t Qs[64 * 72];    // 64 rows x 64 d (+8 pad)
    __shared__ bf16_t Ks[32 * 72];    // 32 keys x 64 d
    __shared__ bf16_t Ps[64 * 40];    // 64 rows x 32 keys
    __shared__ bf16_t Vst[512 * 40];  // transposed: 512 vcols x 32 keys
    __shared__ float  rowscale[64];

    const int tid = threadIdx.x, lane = tid & 31, wid = tid >> 5;
    const int hb = lane >> 4, mm = lane & 15;
    const int qb = blockIdx.x, h = blockIdx.y, b = blockIdx.z;
    const size_t row0q = (size_t)b * SEQ + (size_t)qb * 64;

    // stage Q block (64 x 64) via async DMA
#pragma unroll
    for (int it = 0; it < 2; ++it) {
        int ch = tid + it * 256;
        int r = ch >> 3, d = (ch & 7) * 8;
        async_load_b128(&qkv[(row0q + r) * QKVD + h * HEAD_C + d],
                        lds_off(&Qs[r * 72 + d]));
    }
    wait_async();
    __syncthreads();

    v8f acc[4][4];
#pragma unroll
    for (int r = 0; r < 4; ++r)
#pragma unroll
        for (int c = 0; c < 4; ++c) {
            v8f z = {0.f, 0.f, 0.f, 0.f, 0.f, 0.f, 0.f, 0.f};
            acc[r][c] = z;
        }
    float mrow[8], lrow[8];
#pragma unroll
    for (int i = 0; i < 8; ++i) { mrow[i] = -INFINITY; lrow[i] = 0.0f; }

    v16bf aq[2];
    if (wid < 4) {
#pragma unroll
        for (int s = 0; s < 2; ++s) {
            const bf16_t* p = &Qs[(wid * 16 + mm) * 72 + s * 32 + hb * 16];
            aq[s] = ldfrag(p, p + 8);
        }
    }

    // per-thread staging geometry:
    //  K: one 16B chunk/thread.  V: 8 consecutive keys x 8 cols per thread,
    //  transposed 8x8 in registers, stored as 8x ds_store_b128.
    const int g8 = (tid & 63) * 8;    // V column-group base (8 cols)
    const int o8 = (tid >> 6) * 8;    // V key-octet base (8 keys)
    const bf16_t* pK = &qkv[((size_t)b * SEQ + (tid >> 3)) * QKVD + 512 +
                            h * HEAD_C + (tid & 7) * 8];
    const bf16_t* pV = &qkv[((size_t)b * SEQ + o8) * QKVD + 1024 +
                            h * IN_C + g8];
    const unsigned lK = lds_off(&Ks[(tid >> 3) * 72 + (tid & 7) * 8]);

    for (int kb0 = 0; kb0 < SEQ; kb0 += 32) {
        // stage K block (32 x 64) via async DMA: exactly 256 chunks
        async_load_b128(pK, lK);
        // stage V block: load 8 keys x 8 cols, transpose in regs, b128 stores
        {
            v8bf rr[8];
#pragma unroll
            for (int j = 0; j < 8; ++j)
                rr[j] = *(const v8bf*)(pV + (size_t)j * QKVD);
#pragma unroll
            for (int c = 0; c < 8; ++c) {
                v8bf outv;
#pragma unroll
                for (int j = 0; j < 8; ++j) outv[j] = rr[j][c];
                *(v8bf*)&Vst[(g8 + c) * 40 + o8] = outv;
            }
        }
        pK += (size_t)32 * QKVD;
        pV += (size_t)32 * QKVD;
        wait_async();
        __syncthreads();

        if (wid < 4) {
            // S = Q @ K^T : 16 rows x 32 keys per wave
            v8f sc[2];
#pragma unroll
            for (int c = 0; c < 2; ++c) {
                v8f z = {0.f, 0.f, 0.f, 0.f, 0.f, 0.f, 0.f, 0.f};
                sc[c] = z;
            }
#pragma unroll
            for (int s = 0; s < 2; ++s) {
#pragma unroll
                for (int c = 0; c < 2; ++c) {
                    const bf16_t* p = &Ks[(c * 16 + mm) * 72 + s * 32 + hb * 16];
                    v16bf bk = ldfrag(p, p + 8);
                    sc[c] = wmma_bf16(aq[s], bk, sc[c]);
                }
            }
            // online softmax per row (row reduction across 16-lane halves)
#pragma unroll
            for (int i = 0; i < 8; ++i) {
                float s0 = sc[0][i] * 0.125f;   // 1/sqrt(64)
                float s1 = sc[1][i] * 0.125f;
                float mx = fmaxf(s0, s1);
#pragma unroll
                for (int d = 1; d < 16; d <<= 1)
                    mx = fmaxf(mx, __shfl_xor(mx, d, 32));
                float mnew = fmaxf(mrow[i], mx);
                float alpha = __expf(mrow[i] - mnew);
                float p0 = __expf(s0 - mnew);
                float p1 = __expf(s1 - mnew);
                float rs = p0 + p1;
#pragma unroll
                for (int d = 1; d < 16; d <<= 1)
                    rs += __shfl_xor(rs, d, 32);
                lrow[i] = alpha * lrow[i] + rs;
                mrow[i] = mnew;
                int rowl = wid * 16 + i + hb * 8;
                Ps[rowl * 40 + mm]      = (bf16_t)p0;
                Ps[rowl * 40 + 16 + mm] = (bf16_t)p1;
                if (mm == 0) rowscale[rowl] = alpha;
            }
        }
        __syncthreads();

        // rescale accumulators, then acc += P(64x32) @ V(32x64-chunk)
#pragma unroll
        for (int r = 0; r < 4; ++r)
#pragma unroll
            for (int i = 0; i < 8; ++i) {
                float a = rowscale[r * 16 + i + hb * 8];
#pragma unroll
                for (int c = 0; c < 4; ++c) acc[r][c][i] *= a;
            }

        v16bf bv[4];
#pragma unroll
        for (int c = 0; c < 4; ++c) {
            const bf16_t* pv = &Vst[(wid * 64 + c * 16 + mm) * 40 + hb * 16];
            bv[c] = ldfrag(pv, pv + 8);
        }
#pragma unroll
        for (int r = 0; r < 4; ++r) {
            const bf16_t* pp = &Ps[(r * 16 + mm) * 40 + hb * 16];
            v16bf ap = ldfrag(pp, pp + 8);
#pragma unroll
            for (int c = 0; c < 4; ++c)
                acc[r][c] = wmma_bf16(ap, bv[c], acc[r][c]);
        }
        __syncthreads();
    }

    // finalize: divide by row sums, write bf16 to attn_out (col = h*512 + d)
    if (wid < 4) {
#pragma unroll
        for (int i = 0; i < 8; ++i)
            if (mm == 0) rowscale[wid * 16 + i + hb * 8] = 1.0f / lrow[i];
    }
    __syncthreads();
#pragma unroll
    for (int r = 0; r < 4; ++r)
#pragma unroll
        for (int i = 0; i < 8; ++i) {
            int rowl = r * 16 + i + hb * 8;
            float inv = rowscale[rowl];
            size_t grow = row0q + rowl;
#pragma unroll
            for (int c = 0; c < 4; ++c) {
                int gcol = h * IN_C + wid * 64 + c * 16 + mm;
                attn_out[grow * (size_t)(NHEAD * IN_C) + gcol] =
                    (bf16_t)(acc[r][c][i] * inv);
            }
        }
}

// ---------------------------------------------------------------------------
// Host-side orchestration
// ---------------------------------------------------------------------------
extern "C" void kernel_launch(void* const* d_in, const int* in_sizes, int n_in,
                              void* d_out, int out_size, void* d_ws, size_t ws_size,
                              hipStream_t stream) {
    (void)in_sizes; (void)n_in; (void)out_size; (void)ws_size;
    const float* x       = (const float*)d_in[0];
    const float* t       = (const float*)d_in[1];
    const float* ln1_g   = (const float*)d_in[2];
    const float* ln1_b   = (const float*)d_in[3];
    const float* qkv_w   = (const float*)d_in[4];
    const float* qkv_b   = (const float*)d_in[5];
    const float* merge_w = (const float*)d_in[6];
    const float* merge_b = (const float*)d_in[7];
    const float* time_w  = (const float*)d_in[8];
    const float* time_b  = (const float*)d_in[9];
    const float* ln2_g   = (const float*)d_in[10];
    const float* ln2_b   = (const float*)d_in[11];
    const float* ff1_w   = (const float*)d_in[12];
    const float* ff1_b   = (const float*)d_in[13];
    const float* ff2_w   = (const float*)d_in[14];
    const float* ff2_b   = (const float*)d_in[15];
    float* out = (float*)d_out;

    char* ws = (char*)d_ws;
    size_t off = 0;
    auto take = [&](size_t bytes) -> char* {
        char* p = ws + off;
        off += (bytes + 255) & ~(size_t)255;
        return p;
    };
    bf16_t* qkv_wt   = (bf16_t*)take((size_t)QKVD * IN_C * 2);
    bf16_t* merge_wt = (bf16_t*)take((size_t)IN_C * 4096 * 2);
    bf16_t* ff1_wt   = (bf16_t*)take((size_t)EXPAND_C * 768 * 2);
    bf16_t* ff2_wt   = (bf16_t*)take((size_t)OUT_CH * EXPAND_C * 2);
    bf16_t* x_ln     = (bf16_t*)take((size_t)ROWS * IN_C * 2);
    bf16_t* qkvb     = (bf16_t*)take((size_t)ROWS * QKVD * 2);
    bf16_t* attn_o   = (bf16_t*)take((size_t)ROWS * 4096 * 2);
    float*  x1       = (float*) take((size_t)ROWS * IN_C * 4);
    float*  ttb      = (float*) take((size_t)BATCH * TIME_C * 4);
    bf16_t* h_in     = (bf16_t*)take((size_t)ROWS * 768 * 2);
    bf16_t* hmid     = (bf16_t*)take((size_t)ROWS * EXPAND_C * 2);

    auto cv = [&](const float* W, bf16_t* Wt, int K, int N) {
        size_t n = (size_t)K * (size_t)N;
        convert_wt<<<dim3((unsigned)((n + 255) / 256)), 256, 0, stream>>>(W, Wt, K, N);
    };
    cv(qkv_w,   qkv_wt,   IN_C,     QKVD);
    cv(merge_w, merge_wt, 4096,     IN_C);
    cv(ff1_w,   ff1_wt,   768,      EXPAND_C);
    cv(ff2_w,   ff2_wt,   EXPAND_C, OUT_CH);

    // 1) LN1 -> bf16
    ln1_kernel<<<ROWS, 256, 0, stream>>>(x, ln1_g, ln1_b, x_ln);
    // 2) QKV GEMM: (8192x512)@(512x5120) + bias -> bf16
    gemm_bf16<0, 0, 0><<<dim3(QKVD / 128, ROWS / 128), 256, 0, stream>>>(
        x_ln, qkv_wt, qkv_b, nullptr, qkvb, ROWS, QKVD, IN_C);
    // 3) Flash attention -> (8192 x 4096) bf16
    flash_attn<<<dim3(SEQ / 64, NHEAD, BATCH), 256, 0, stream>>>(qkvb, attn_o);
    // 4) Merge GEMM + residual x : (8192x4096)@(4096x512) -> f32 x1
    gemm_bf16<0, 1, 1><<<dim3(IN_C / 128, ROWS / 128), 256, 0, stream>>>(
        attn_o, merge_wt, merge_b, x, x1, ROWS, IN_C, 4096);
    // 5) time MLP (tiny)
    time_kernel<<<BATCH, 256, 0, stream>>>(t, time_w, time_b, ttb);
    // 6) LN2 over concat(x1, tt) -> bf16 (8192 x 768)
    ln2_kernel<<<ROWS, 256, 0, stream>>>(x1, ttb, ln2_g, ln2_b, h_in);
    // 7) FF1 + GELU: (8192x768)@(768x2048) -> bf16
    gemm_bf16<1, 0, 0><<<dim3(EXPAND_C / 128, ROWS / 128), 256, 0, stream>>>(
        h_in, ff1_wt, ff1_b, nullptr, hmid, ROWS, EXPAND_C, 768);
    // 8) FF2 + residual x1: (8192x2048)@(2048x512) -> f32 d_out
    gemm_bf16<0, 1, 1><<<dim3(OUT_CH / 128, ROWS / 128), 256, 0, stream>>>(
        hmid, ff2_wt, ff2_b, x1, out, ROWS, OUT_CH, EXPAND_C);
}
